// MultiHeadAttention_8289286881623
// MI455X (gfx1250) — matrix operand order repaired and checked by reading, not compiled, not run on hardware
//
#include <hip/hip_runtime.h>

// ---------------- types ----------------
typedef __attribute__((ext_vector_type(16))) __bf16 v16bf;
typedef __attribute__((ext_vector_type(8)))  __bf16 v8bf;
typedef __attribute__((ext_vector_type(8)))  float   v8f;
typedef __attribute__((ext_vector_type(4)))  float   v4f;
typedef __attribute__((ext_vector_type(4)))  unsigned u32x4;
typedef __attribute__((ext_vector_type(8)))  int      i32x8;
typedef __attribute__((ext_vector_type(4)))  int      i32x4;

#define DMODEL 1024
#define NHEAD  16
#define DH     64
#define SEQ    2048
#define BATCH  4
#define MROWS  (BATCH * SEQ) // 8192

// fp32 -> bf16, round-to-nearest-even, pure bit ops (no fp-conversion dependence)
__device__ __forceinline__ __bf16 to_bf16(float f) {
    unsigned u = __builtin_bit_cast(unsigned, f);
    unsigned r = u + 0x7FFFu + ((u >> 16) & 1u);
    unsigned short h = (unsigned short)(r >> 16);
    return __builtin_bit_cast(__bf16, h);
}

__device__ __forceinline__ v8f wmma_bf16(v16bf a, v16bf b, v8f c) {
    // D = A(16x32) * B(32x16) + C(16x16), fp32 accum
    return __builtin_amdgcn_wmma_f32_16x16x32_bf16(
        /*neg_a=*/false, a, /*neg_b=*/false, b,
        /*c_mod=*/(short)0, c, /*reuse_a=*/false, /*reuse_b=*/false);
}

// ---------------- fragment loaders ----------------
// A fragment 16x32 bf16: lane L<16 holds row M=L, K = k0..k0+15 contiguous;
// lane L>=16 holds row M=L-16, K = k0+16..k0+31.
__device__ __forceinline__ v16bf load_a(const __bf16* __restrict__ A, int lda,
                                        int row0, int k0, int lane) {
    const int r  = row0 + (lane & 15);
    const int kk = k0 + ((lane >> 4) << 4);
    const __bf16* p = A + (long)r * lda + kk;
    v8bf lo = *(const v8bf*)(p);        // K = kk .. kk+7
    v8bf hi = *(const v8bf*)(p + 8);    // K = kk+8 .. kk+15
    v16bf o;
#pragma unroll
    for (int i = 0; i < 8; i++) { o[i] = lo[i]; o[i + 8] = hi[i]; }
    return o;
}

// Same fragment from fp32 source, converting in-register.
__device__ __forceinline__ v16bf load_a(const float* __restrict__ A, int lda,
                                        int row0, int k0, int lane) {
    const int r  = row0 + (lane & 15);
    const int kk = k0 + ((lane >> 4) << 4);
    const float* p = A + (long)r * lda + kk;
    v4f a0 = *(const v4f*)(p);
    v4f a1 = *(const v4f*)(p + 4);
    v4f b0 = *(const v4f*)(p + 8);
    v4f b1 = *(const v4f*)(p + 12);
    v16bf o;
#pragma unroll
    for (int i = 0; i < 4; i++) {
        o[i]      = to_bf16(a0[i]);
        o[i + 4]  = to_bf16(a1[i]);
        o[i + 8]  = to_bf16(b0[i]);
        o[i + 12] = to_bf16(b1[i]);
    }
    return o;
}

// B fragment 32x16 bf16: lane L<16 holds column N=L, K = k0..k0+15 contiguous;
// lane L>=16 holds column N=L-16, K = k0+16..k0+31. BT is column-major B (row n = column n).
__device__ __forceinline__ v16bf load_b(const __bf16* __restrict__ BT, int ldb,
                                        int col0, int k0, int lane) {
    const int n  = col0 + (lane & 15);
    const int kk = k0 + ((lane >> 4) << 4);
    return *(const v16bf*)(BT + (long)n * ldb + kk);
}

// ---------------- TDM: DMA a [64 rows][32 cols] bf16 slab (rows strided 2048) into LDS ----------
// D# per CDNA5 ISA: group0 = {count=1, lds_addr, global_addr, type=2};
// group1 = {data_size=2B, tensor_dim0=2048, tensor_dim1=64, tile_dim0=32, tile_dim1=64,
//           tensor_dim0_stride=2048}; groups 2/3 zero (2-D tile).
__device__ __forceinline__ void tdm_load_vslab(const __bf16* gsrc, unsigned lds_off) {
    unsigned long long ga = (unsigned long long)(uintptr_t)gsrc;
    u32x4 g0;
    g0[0] = 1u;                                               // count=1, user descriptor
    g0[1] = lds_off;                                          // lds_addr (bytes)
    g0[2] = (unsigned)(ga & 0xFFFFFFFFu);                     // global_addr[31:0]
    g0[3] = (unsigned)((ga >> 32) & 0x01FFFFFFu) | (2u << 30);// global_addr[56:32] | type=2
    i32x8 g1;
    g1[0] = 1 << 16;                  // data_size = 1 (2 bytes)
    g1[1] = (int)(2048u << 16);       // tensor_dim0 = 2048 (low16 -> bits 63:48)
    g1[2] = 64 << 16;                 // tensor_dim1 = 64 (low16 -> bits 95:80)
    g1[3] = 32 << 16;                 // tile_dim0 = 32 (bits 127:112)
    g1[4] = 64;                       // tile_dim1 = 64 (bits 143:128), tile_dim2 = 0
    g1[5] = 2048;                     // tensor_dim0_stride = 2048 (low32)
    g1[6] = 0;                        // stride hi / tensor_dim1_stride lo
    g1[7] = 0;
    i32x4 gz = {};
#if defined(__clang_major__) && (__clang_major__ >= 23)
    i32x8 gz8 = {};
    __builtin_amdgcn_tensor_load_to_lds(g0, g1, gz, gz, gz8, 0);
#else
    __builtin_amdgcn_tensor_load_to_lds(g0, g1, gz, gz, 0);
#endif
}

// ---------------- generic GEMM: C = A(MxK) * BT^T + bias ----------------
// 128x128 block tile, 8 waves as 2(m) x 4(n), wave tile 64x32 = 4x2 WMMA tiles.
template <typename AT, bool OUT_BF16>
__global__ __launch_bounds__(256) void gemm_k(const AT* __restrict__ A, int lda,
                                              const __bf16* __restrict__ BT, int ldb,
                                              const float* __restrict__ bias,
                                              void* __restrict__ Cout, int ldc, int Kdim) {
    const int lane = threadIdx.x & 31;
    const int wave = threadIdx.x >> 5;
    const int m_wave = blockIdx.y * 128 + (wave & 1) * 64;
    const int n_wave = blockIdx.x * 128 + (wave >> 1) * 32;

    const v8f vzero = {};
    v8f acc[4][2];
#pragma unroll
    for (int i = 0; i < 4; i++)
#pragma unroll
        for (int j = 0; j < 2; j++) acc[i][j] = vzero;

    for (int kk = 0; kk < Kdim; kk += 32) {
        if (kk + 64 < Kdim) { // prefetch next-next K-slab -> global_prefetch_b8
            __builtin_prefetch(&A[(long)(m_wave + (lane & 15)) * lda + kk + 64], 0, 1);
            __builtin_prefetch(&BT[(long)(n_wave + (lane & 15)) * ldb + kk + 64], 0, 1);
        }
        v16bf bfr[2];
#pragma unroll
        for (int j = 0; j < 2; j++) bfr[j] = load_b(BT, ldb, n_wave + j * 16, kk, lane);
#pragma unroll
        for (int i = 0; i < 4; i++) {
            v16bf afr = load_a(A, lda, m_wave + i * 16, kk, lane);
#pragma unroll
            for (int j = 0; j < 2; j++) acc[i][j] = wmma_bf16(afr, bfr[j], acc[i][j]);
        }
    }

    // C/D layout: VGPR r, lanes 0-15 -> (M=r, N=lane), lanes 16-31 -> (M=r+8, N=lane-16)
    const int rbase = (lane >> 4) << 3;
    const int cl = lane & 15;
#pragma unroll
    for (int i = 0; i < 4; i++)
#pragma unroll
        for (int j = 0; j < 2; j++) {
            const int col = n_wave + j * 16 + cl;
            const float bz = bias[col];
#pragma unroll
            for (int r = 0; r < 8; r++) {
                const int row = m_wave + i * 16 + rbase + r;
                const float val = acc[i][j][r] + bz;
                if constexpr (OUT_BF16)
                    ((__bf16*)Cout)[(long)row * ldc + col] = to_bf16(val);
                else
                    ((float*)Cout)[(long)row * ldc + col] = val;
            }
        }
}

// ---------------- scores: S = scale * Qh @ Kh^T (per b,h), fp32 pre-softmax ----------------
__global__ __launch_bounds__(256) void scores_k(const __bf16* __restrict__ Qp,
                                                const __bf16* __restrict__ Kp,
                                                float* __restrict__ Attn, float scale) {
    const int lane = threadIdx.x & 31;
    const int wave = threadIdx.x >> 5;
    const int z = blockIdx.z;
    const int b = z >> 4, h = z & 15;
    const __bf16* Qb = Qp + (long)b * SEQ * DMODEL + h * DH;
    const __bf16* Kb = Kp + (long)b * SEQ * DMODEL + h * DH;
    float* Ab = Attn + (long)z * SEQ * SEQ;

    const int m_wave = blockIdx.y * 128 + (wave & 1) * 64;
    const int n_wave = blockIdx.x * 128 + (wave >> 1) * 32;

    const v8f vzero = {};
    v8f acc[4][2];
#pragma unroll
    for (int i = 0; i < 4; i++)
#pragma unroll
        for (int j = 0; j < 2; j++) acc[i][j] = vzero;

#pragma unroll
    for (int kk = 0; kk < DH; kk += 32) {
        v16bf bfr[2];
#pragma unroll
        for (int j = 0; j < 2; j++) bfr[j] = load_b(Kb, DMODEL, n_wave + j * 16, kk, lane);
#pragma unroll
        for (int i = 0; i < 4; i++) {
            v16bf afr = load_a(Qb, DMODEL, m_wave + i * 16, kk, lane);
#pragma unroll
            for (int j = 0; j < 2; j++) acc[i][j] = wmma_bf16(afr, bfr[j], acc[i][j]);
        }
    }

    const int rbase = (lane >> 4) << 3;
    const int cl = lane & 15;
#pragma unroll
    for (int i = 0; i < 4; i++)
#pragma unroll
        for (int j = 0; j < 2; j++) {
            const int col = n_wave + j * 16 + cl;
#pragma unroll
            for (int r = 0; r < 8; r++) {
                const int row = m_wave + i * 16 + rbase + r;
                Ab[(long)row * SEQ + col] = acc[i][j][r] * scale;
            }
        }
}

// ---------------- softmax: one 256-thread block per 2048-wide row, in place ----------------
__global__ __launch_bounds__(256) void softmax_k(float* __restrict__ Attn) {
    __shared__ float red[256];
    const int t = threadIdx.x;
    float* p = Attn + (long)blockIdx.x * SEQ;

    float x[8];
    float m = -3.4e38f;
#pragma unroll
    for (int i = 0; i < 8; i++) { x[i] = p[t + i * 256]; m = fmaxf(m, x[i]); }
    red[t] = m;
    __syncthreads();
    for (int s = 128; s > 0; s >>= 1) {
        if (t < s) red[t] = fmaxf(red[t], red[t + s]);
        __syncthreads();
    }
    m = red[0];
    __syncthreads();

    float sum = 0.f;
#pragma unroll
    for (int i = 0; i < 8; i++) { x[i] = __expf(x[i] - m); sum += x[i]; }
    red[t] = sum;
    __syncthreads();
    for (int s = 128; s > 0; s >>= 1) {
        if (t < s) red[t] += red[t + s];
        __syncthreads();
    }
    const float inv = 1.f / red[0];
#pragma unroll
    for (int i = 0; i < 8; i++) p[t + i * 256] = x[i] * inv;
}

// ---------------- ctx: Ctx_h = P @ Vh, per (b,h); P fp32 -> bf16 in-register ----------------
// V K-slab (32x64, 4KB) staged into LDS by the Tensor Data Mover, double-buffered; all 8
// waves read B-fragments from LDS (8x traffic reduction vs per-wave global fragments).
__global__ __launch_bounds__(256) void ctx_k(const float* __restrict__ Attn,
                                             const __bf16* __restrict__ VpT,
                                             __bf16* __restrict__ Ctx) {
    __shared__ __attribute__((aligned(32))) __bf16 vtile[2][DH * 32]; // [buf][n(64)][k(32)]
    const int lane = threadIdx.x & 31;
    const int wave = threadIdx.x >> 5;
    const int z = blockIdx.y, b = z >> 4, h = z & 15;
    const float* Pb = Attn + (long)z * SEQ * SEQ;
    const __bf16* Vb = VpT + (long)z * DH * SEQ; // [dh][SEQ] row-major: row n = V column n
    const int m0 = blockIdx.x * 128 + wave * 16;

    const v8f vzero = {};
    v8f acc[4];
#pragma unroll
    for (int j = 0; j < 4; j++) acc[j] = vzero;

    if (wave == 0)
        tdm_load_vslab(Vb, (unsigned)(uintptr_t)&vtile[0][0]);

    const int nsteps = SEQ / 32;
    for (int s = 0; s < nsteps; s++) {
        const int kk = s * 32;
        const int cur = s & 1;
        if (wave == 0) __builtin_amdgcn_s_wait_tensorcnt(0); // slab `cur` landed in LDS
        __syncthreads();                                     // visible to all waves
        if (wave == 0 && (s + 1) < nsteps)                   // overlap next DMA with compute
            tdm_load_vslab(Vb + kk + 32, (unsigned)(uintptr_t)&vtile[1 - cur][0]);

        if (kk + 64 < SEQ)
            __builtin_prefetch(&Pb[(long)(m0 + (lane & 15)) * SEQ + kk + 64], 0, 1);
        v16bf afr = load_a(Pb, SEQ, m0, kk, lane);
        const int chunk = (lane >> 4) << 4; // K-half per B-fragment rules
#pragma unroll
        for (int j = 0; j < 4; j++) {
            const int n = j * 16 + (lane & 15);
            v16bf bfr = *(const v16bf*)&vtile[cur][n * 32 + chunk]; // ds_load 32B contiguous
            acc[j] = wmma_bf16(afr, bfr, acc[j]);
        }
    }

    const int rbase = (lane >> 4) << 3;
    const int cl = lane & 15;
#pragma unroll
    for (int j = 0; j < 4; j++) {
        const int col = h * DH + j * 16 + cl;
#pragma unroll
        for (int r = 0; r < 8; r++) {
            const int row = b * SEQ + m0 + rbase + r;
            Ctx[(long)row * DMODEL + col] = to_bf16(acc[j][r]);
        }
    }
}

// ---------------- small data-movement kernels ----------------
// W [K=1024][N=1024] fp32 -> WT [N][K] bf16 (coalesced writes)
__global__ void wtrans_k(const float* __restrict__ W, __bf16* __restrict__ WT) {
    const int idx = blockIdx.x * 256 + threadIdx.x; // 1M
    const int n = idx >> 10, k = idx & 1023;
    WT[(long)n * 1024 + k] = to_bf16(W[(long)k * 1024 + n]);
}

// Vp [B*SEQ][DMODEL] bf16 -> VpT [(b*16+h)*64+n][SEQ] bf16 (coalesced writes)
__global__ void vtrans_k(const __bf16* __restrict__ Vp, __bf16* __restrict__ VpT) {
    const long idx = (long)blockIdx.x * 256 + threadIdx.x; // 8M
    const int kpos = (int)(idx & 2047);
    const long rest = idx >> 11;
    const int n = (int)(rest & 63);
    const int zh = (int)(rest >> 6);
    const int h = zh & 15, b = zh >> 4;
    VpT[idx] = Vp[((long)b * SEQ + kpos) * DMODEL + h * DH + n];
}

// ---------------- launcher ----------------
extern "C" void kernel_launch(void* const* d_in, const int* in_sizes, int n_in,
                              void* d_out, int out_size, void* d_ws, size_t ws_size,
                              hipStream_t stream) {
    const float* q  = (const float*)d_in[0];
    const float* k  = (const float*)d_in[1];
    const float* v  = (const float*)d_in[2];
    const float* Wq = (const float*)d_in[3];
    const float* bq = (const float*)d_in[4];
    const float* Wk = (const float*)d_in[5];
    const float* bk = (const float*)d_in[6];
    const float* Wv = (const float*)d_in[7];
    const float* bv = (const float*)d_in[8];
    const float* Wo = (const float*)d_in[9];
    const float* bo = (const float*)d_in[10];

    float* out  = (float*)d_out;                     // [8192,1024]
    float* attn = out + (long)MROWS * DMODEL;        // [4,16,2048,2048]

    // workspace (bf16), ~88 MiB total
    __bf16* ws = (__bf16*)d_ws;
    const long WSZ = (long)1024 * 1024;   // per transposed weight
    const long XSZ = (long)MROWS * DMODEL;
    __bf16* WqT = ws;
    __bf16* WkT = WqT + WSZ;
    __bf16* WvT = WkT + WSZ;
    __bf16* WoT = WvT + WSZ;
    __bf16* Qp  = WoT + WSZ;
    __bf16* Kp  = Qp + XSZ;
    __bf16* Vp  = Kp + XSZ;
    __bf16* VpT = Vp + XSZ;
    __bf16* Ctx = VpT + XSZ;

    dim3 blk(256);

    wtrans_k<<<4096, blk, 0, stream>>>(Wq, WqT);
    wtrans_k<<<4096, blk, 0, stream>>>(Wk, WkT);
    wtrans_k<<<4096, blk, 0, stream>>>(Wv, WvT);
    wtrans_k<<<4096, blk, 0, stream>>>(Wo, WoT);

    dim3 gg(DMODEL / 128, MROWS / 128); // (8, 64)
    gemm_k<float, true><<<gg, blk, 0, stream>>>(q, DMODEL, WqT, DMODEL, bq, Qp, DMODEL, DMODEL);
    gemm_k<float, true><<<gg, blk, 0, stream>>>(k, DMODEL, WkT, DMODEL, bk, Kp, DMODEL, DMODEL);
    gemm_k<float, true><<<gg, blk, 0, stream>>>(v, DMODEL, WvT, DMODEL, bv, Vp, DMODEL, DMODEL);

    vtrans_k<<<32768, blk, 0, stream>>>(Vp, VpT);

    // scale = 1/sqrt(B) = 0.5 (reference quirk, reproduced)
    scores_k<<<dim3(SEQ / 128, SEQ / 128, BATCH * NHEAD), blk, 0, stream>>>(Qp, Kp, attn, 0.5f);
    softmax_k<<<BATCH * NHEAD * SEQ, blk, 0, stream>>>(attn);
    ctx_k<<<dim3(SEQ / 128, BATCH * NHEAD), blk, 0, stream>>>(attn, VpT, Ctx);
    gemm_k<__bf16, false><<<gg, blk, 0, stream>>>(Ctx, DMODEL, WoT, DMODEL, bo, out, DMODEL, DMODEL);
}
